// ProbProtoSegHead_1236950581879
// MI455X (gfx1250) — matrix operand order repaired and checked by reading, not compiled, not run on hardware
//
#include <hip/hip_runtime.h>
#include <hip/hip_bf16.h>

// ---------------------------------------------------------------------------
// ProbProtoSegHead for MI455X (gfx1250, wave32, WMMA).
//
// dot/cross GEMMs: M=131072 pixels, N=190(->192) proto cols, K=720.
// 72 GFLOP vs 765 MB HBM traffic -> AI ~94 FLOP/B; only the bf16 WMMA path
// (8x the MACs/issue of f32 16x16x4) approaches the 23.3 TB/s roofline, so
// A/B are bf16 with f32 accumulation. Prototypes (540 KB) are L2-resident;
// a prep kernel l2-normalizes them and packs both proto operands into the
// exact 16-bit WMMA A-matrix register image (coalesced b128 tile loads).
//
// v3: bf16 pair packing via v_perm_b32 (__builtin_amdgcn_perm) with +0x8000
// round bias: 3 VALU/pair instead of the 5-op manual RNE sequence seen in
// the v2 disasm (cvt_pk_bf16 builtin is not declared on this toolchain).
// Keeps v2's pipelining: 12 A-tiles batched per K-step (clause + incremental
// loadcnt waits) and B operand loads prefetched one K-tile ahead.
// ---------------------------------------------------------------------------

#define NUM_CLASSES 19
#define NUM_PROTO   10
#define PROJ_DIM    720
#define NROWS       192           // 190 proto rows padded to 12*16
#define JT          12            // M-tiles of 16
#define KT          23            // ceil(720/32) K-tiles (last is half)
#define HWP         32768         // H*W = 128*256
#define NPIX        131072        // B*H*W
#define LN_EPS      1e-5f
#define LAMDA       (1.0f / 720.0f)
#define APK_U32     (JT * KT * 32 * 8)   // 70656 u32 per packed operand

typedef __attribute__((ext_vector_type(16))) __bf16 bf16x16;
typedef __attribute__((ext_vector_type(8)))  float  v8f;

union Pack8 {
    uint4    q[2];
    unsigned u[8];
    bf16x16  v;
};

// f32x2 -> packed bf16x2, round-half-up (1 ulp-tie difference vs RNE).
// v_add_u32 x2 + v_perm_b32: dst = { hi[31:16], lo[31:16] }.
__device__ __forceinline__ unsigned pack_bf16(float lo, float hi) {
    unsigned a = __builtin_bit_cast(unsigned, lo) + 0x8000u;
    unsigned b = __builtin_bit_cast(unsigned, hi) + 0x8000u;
    // v_perm_b32: byte sel 0-3 -> src1 (a), 4-7 -> src0 (b)
    return __builtin_amdgcn_perm(b, a, 0x07060302u);
}

// --------------------------- prep kernel 1 ---------------------------------
__global__ void prep_rows(const float* __restrict__ protos,
                          const float* __restrict__ pvar,
                          float* __restrict__ rnorm,
                          float* __restrict__ pv_sum) {
    int j = threadIdx.x;
    if (j >= NROWS) return;
    if (j < NUM_CLASSES * NUM_PROTO) {
        const float* pr = protos + (size_t)j * PROJ_DIM;
        const float* vr = pvar   + (size_t)j * PROJ_DIM;
        float s2 = 0.f, pv = 0.f;
        for (int k = 0; k < PROJ_DIM; ++k) {
            float p = pr[k];
            s2 += p * p;
            pv += vr[k];
        }
        rnorm[j]  = 1.0f / fmaxf(sqrtf(s2), 1e-12f);
        pv_sum[j] = pv;
    } else {
        rnorm[j]  = 0.f;
        pv_sum[j] = 0.f;
    }
}

// --------------------------- prep kernel 2 ---------------------------------
// Pack l2-normalized protos (bf16) and sqrt(proto_var) (bf16) into the
// 16x32 16-bit WMMA A-matrix lane/VGPR layout (contiguous k = 16h + s, same order as the B packing):
//   lane l<16 : row M=l,  VGPR g holds K = 2g, +1
//   lane l>=16: row M=l-16, same but K offset +16
// Index: [mtile][ktile][lane][g], 8 consecutive u32 per lane -> b128 x2.
__global__ void prep_pack(const float* __restrict__ protos,
                          const float* __restrict__ pvar,
                          const float* __restrict__ rnorm,
                          unsigned* __restrict__ Adot,
                          unsigned* __restrict__ Avar) {
    int t = blockIdx.x * blockDim.x + threadIdx.x;
    if (t >= JT * KT * 32) return;
    int lane  = t & 31;
    int tile  = t >> 5;                 // tile = jt*KT + kt
    int kt    = tile % KT;
    int j     = (tile / KT) * 16 + (lane & 15);
    int lhalf = lane >> 4;
    bool jv   = j < NUM_CLASSES * NUM_PROTO;
    float rn  = jv ? rnorm[j] : 0.f;
    const float* pr = protos + (size_t)j * PROJ_DIM;
    const float* vr = pvar   + (size_t)j * PROJ_DIM;
    size_t base = ((size_t)tile * 32 + lane) * 8;
#pragma unroll
    for (int g = 0; g < 8; ++g) {
        int koff = lhalf * 16 + g * 2;
        int k = kt * 32 + koff;
        float a0 = 0.f, a1 = 0.f, s0 = 0.f, s1 = 0.f;
        if (jv && k < PROJ_DIM)     { a0 = pr[k]   * rn; s0 = sqrtf(vr[k]);   }
        if (jv && k + 1 < PROJ_DIM) { a1 = pr[k+1] * rn; s1 = sqrtf(vr[k+1]); }
        Adot[base + g] = pack_bf16(a0, a1);
        Avar[base + g] = pack_bf16(s0, s1);
    }
}

// ------------------------- operand tile loaders ----------------------------
// B operand from x [C,H,W]: per-k-plane strided b32 loads (coalesced 64B per
// lane half; neighbor waves consume the other half of each cache line).
__device__ __forceinline__ void load_x_tile(const float* __restrict__ xb,
                                            int kt, int lhalf,
                                            float* lo, float* hi) {
    int kb = kt * 32 + lhalf * 16;
#pragma unroll
    for (int g = 0; g < 8; ++g) {
        int k = kb + 2 * g;               // even; k+1 valid whenever k is
        float l = 0.f, h = 0.f;
        if (k < PROJ_DIM) {
            l = xb[(size_t)k * HWP];
            h = xb[(size_t)(k + 1) * HWP];
        }
        lo[g] = l;
        hi[g] = h;
    }
}

// B operand from x_var [N,720]: row-major -> 4x b128 per lane.
__device__ __forceinline__ void load_var_tile(const float* __restrict__ xvb,
                                              int kt, int lhalf, float* vals) {
    int kb = kt * 32 + lhalf * 16;
    if (kb < PROJ_DIM) {
        const float4* q4 = (const float4*)(xvb + kb);
        float4 q0 = q4[0], q1 = q4[1], q2 = q4[2], q3 = q4[3];
        vals[0]=q0.x;  vals[1]=q0.y;  vals[2]=q0.z;  vals[3]=q0.w;
        vals[4]=q1.x;  vals[5]=q1.y;  vals[6]=q1.z;  vals[7]=q1.w;
        vals[8]=q2.x;  vals[9]=q2.y;  vals[10]=q2.z; vals[11]=q2.w;
        vals[12]=q3.x; vals[13]=q3.y; vals[14]=q3.z; vals[15]=q3.w;
    } else {
#pragma unroll
        for (int i = 0; i < 16; ++i) vals[i] = 0.f;
    }
}

// ----------------------------- main kernel ---------------------------------
// 4 waves / block; each wave owns 16 pixels x all 192 proto columns.
__global__ __launch_bounds__(128)
void proto_head_main(const float* __restrict__ x,       // [4,720,128,256]
                     const float* __restrict__ x_var,   // [131072,720]
                     const unsigned* __restrict__ Adot, // packed bf16 A
                     const unsigned* __restrict__ Avar, // packed bf16 A
                     const float* __restrict__ pv_sum,  // [192]
                     const float* __restrict__ lnw,     // [19]
                     const float* __restrict__ lnb,     // [19]
                     float* __restrict__ out) {         // [4,19,128,256]
    __shared__ float sims[4][16][200];   // per-wave [pixel][row] scratch
    __shared__ float pvs[NROWS];
    __shared__ float wls[NUM_CLASSES], bls[NUM_CLASSES];

    int tid = threadIdx.x;
    if (tid < NROWS) pvs[tid] = pv_sum[tid];
    if (tid < NUM_CLASSES) { wls[tid] = lnw[tid]; bls[tid] = lnb[tid]; }
    __syncthreads();

    int wv    = tid >> 5;
    int lane  = tid & 31;
    int col   = lane & 15;        // pixel within wave tile / D-matrix column
    int lhalf = lane >> 4;        // which K half / which M half of D
    int n0    = (blockIdx.x * 4 + wv) * 16;
    int bidx  = n0 >> 15;         // / 32768
    int p0    = n0 & (HWP - 1);

    const float* xb = x + (size_t)bidx * PROJ_DIM * HWP + p0 + col;

    v8f acc[JT] = {};
    v8f vzero = {};

    // ============ pass 1: dot = xf @ protos_n^T ============
    {
        float blo[8], bhi[8];
        load_x_tile(xb, 0, lhalf, blo, bhi);   // prologue
        for (int kt = 0; kt < KT; ++kt) {
            Pack8 B;
#pragma unroll
            for (int g = 0; g < 8; ++g) B.u[g] = pack_bf16(blo[g], bhi[g]);
            if (kt + 1 < KT)                    // prefetch next K-tile of x
                load_x_tile(xb, kt + 1, lhalf, blo, bhi);

            Pack8 A[JT];                        // batch: one load clause
#pragma unroll
            for (int mt = 0; mt < JT; ++mt) {
                const uint4* ap =
                    (const uint4*)(Adot + (((size_t)mt * KT + kt) * 32 + lane) * 8);
                A[mt].q[0] = ap[0];
                A[mt].q[1] = ap[1];
            }
#pragma unroll
            for (int mt = 0; mt < JT; ++mt)
                acc[mt] = __builtin_amdgcn_wmma_f32_16x16x32_bf16(
                    false, A[mt].v, false, B.v, (short)0, acc[mt], false, false);
        }
    }

    // park 2*dot in LDS (D layout: lane half selects M+8, col = pixel)
#pragma unroll
    for (int mt = 0; mt < JT; ++mt) {
#pragma unroll
        for (int r = 0; r < 8; ++r) {
            int row = mt * 16 + lhalf * 8 + r;
            sims[wv][col][row] = 2.0f * acc[mt][r];
        }
    }

    // ============ pass 2: cross = sqrt(x_var) @ sqrt(proto_var)^T ==========
#pragma unroll
    for (int mt = 0; mt < JT; ++mt) acc[mt] = vzero;

    float vsum = 0.f;
    {
        const float* xvb = x_var + (size_t)(p0 + col) * PROJ_DIM;
        float vals[16];
        load_var_tile(xvb, 0, lhalf, vals);    // prologue
        for (int kt = 0; kt < KT; ++kt) {
#pragma unroll
            for (int i = 0; i < 16; ++i) vsum += vals[i];
            Pack8 B;
#pragma unroll
            for (int g = 0; g < 8; ++g)
                B.u[g] = pack_bf16(sqrtf(vals[2 * g]), sqrtf(vals[2 * g + 1]));
            if (kt + 1 < KT)                    // prefetch next K-tile of x_var
                load_var_tile(xvb, kt + 1, lhalf, vals);

            Pack8 A[JT];
#pragma unroll
            for (int mt = 0; mt < JT; ++mt) {
                const uint4* ap =
                    (const uint4*)(Avar + (((size_t)mt * KT + kt) * 32 + lane) * 8);
                A[mt].q[0] = ap[0];
                A[mt].q[1] = ap[1];
            }
#pragma unroll
            for (int mt = 0; mt < JT; ++mt)
                acc[mt] = __builtin_amdgcn_wmma_f32_16x16x32_bf16(
                    false, A[mt].v, false, B.v, (short)0, acc[mt], false, false);
        }
    }

    // per-pixel variance sum: lane halves cover complementary K ranges
    float xv_tot = vsum + __shfl_xor(vsum, 16, 32);

    // combine: sim = 2*dot - 2 - lamda*(xv+pv) + 2*lamda*cross
#pragma unroll
    for (int mt = 0; mt < JT; ++mt) {
#pragma unroll
        for (int r = 0; r < 8; ++r) {
            int row = mt * 16 + lhalf * 8 + r;
            float v = sims[wv][col][row];
            v += 2.0f * LAMDA * acc[mt][r] - 2.0f - LAMDA * (xv_tot + pvs[row]);
            sims[wv][col][row] = v;
        }
    }
    __syncthreads();

    // epilogue: max over 10 protos per class, LayerNorm over 19 classes
    if (lane < 16) {
        float cs[NUM_CLASSES];
        float ssum = 0.f;
#pragma unroll
        for (int c = 0; c < NUM_CLASSES; ++c) {
            float m = -3.4e38f;
#pragma unroll
            for (int pm = 0; pm < NUM_PROTO; ++pm)
                m = fmaxf(m, sims[wv][lane][c * NUM_PROTO + pm]);
            cs[c] = m;
            ssum += m;
        }
        float mu = ssum * (1.0f / NUM_CLASSES);
        float vv = 0.f;
#pragma unroll
        for (int c = 0; c < NUM_CLASSES; ++c) {
            float d = cs[c] - mu;
            vv += d * d;
        }
        float rstd = rsqrtf(vv * (1.0f / NUM_CLASSES) + LN_EPS);
        size_t ob = (size_t)bidx * NUM_CLASSES * HWP + p0 + lane;
#pragma unroll
        for (int c = 0; c < NUM_CLASSES; ++c)
            out[ob + (size_t)c * HWP] = (cs[c] - mu) * rstd * wls[c] + bls[c];
    }
}

// ------------------------------- launcher ----------------------------------
extern "C" void kernel_launch(void* const* d_in, const int* in_sizes, int n_in,
                              void* d_out, int out_size, void* d_ws, size_t ws_size,
                              hipStream_t stream) {
    (void)in_sizes; (void)n_in; (void)out_size; (void)ws_size;
    const float* x      = (const float*)d_in[0];   // [4,720,128,256]
    const float* x_var  = (const float*)d_in[1];   // [131072,720]
    const float* protos = (const float*)d_in[2];   // [19,10,720]
    const float* pvar   = (const float*)d_in[3];   // [19,10,720]
    const float* lnw    = (const float*)d_in[4];   // [19]
    const float* lnb    = (const float*)d_in[5];   // [19]
    float*       out    = (float*)d_out;

    unsigned* Adot  = (unsigned*)d_ws;                 // APK_U32 u32
    unsigned* Avar  = Adot + APK_U32;                  // APK_U32 u32
    float*    pvsum = (float*)(Avar + APK_U32);        // 192 f32
    float*    rnorm = pvsum + NROWS;                   // 192 f32

    prep_rows<<<1, NROWS, 0, stream>>>(protos, pvar, rnorm, pvsum);
    int packThreads = JT * KT * 32;                    // 8832
    prep_pack<<<(packThreads + 255) / 256, 256, 0, stream>>>(
        protos, pvar, rnorm, Adot, Avar);
    proto_head_main<<<NPIX / 64, 128, 0, stream>>>(
        x, x_var, Adot, Avar, pvsum, lnw, lnb, out);
}
